// EvoBlock_25469156065565
// MI455X (gfx1250) — matrix operand order repaired and checked by reading, not compiled, not run on hardware
//
#include <hip/hip_runtime.h>
#include <hip/hip_bf16.h>

// ---------------------------------------------------------------------------
// Evoformer-ish block for MI455X (gfx1250): all GEMMs/convs via
// v_wmma_f32_16x16x32_bf16 (bf16 in, fp32 accumulate).
// Heavy convs / big GEMM: register-blocked 2Mx8N tiles per wave ->
// 16 independent WMMA accumulators.  Small GEMMs + pair MLP use dual
// accumulator chains so consecutive WMMAs are independent (no hazard NOPs).
// ---------------------------------------------------------------------------

#define LSEQ   384
#define CSd    384
#define CZd    128
#define NH     8
#define HWd    48
#define HWP    64          // head dim padded to multiple of 32 for WMMA K
#define PAIRN  (384*384)   // 147456

typedef __bf16 bf16_t;
typedef __attribute__((ext_vector_type(16))) __bf16 v16bf;
typedef __attribute__((ext_vector_type(8)))  float  v8f;

// ------------------------------ WMMA helpers -------------------------------

static __device__ __forceinline__ v8f wmma_bf16(v16bf a, v16bf b, v8f c) {
  // D = A(16x32) * B(32x16) + C, fp32 accumulate
  return __builtin_amdgcn_wmma_f32_16x16x32_bf16(
      /*neg_a=*/false, a, /*neg_b=*/false, b,
      /*c_mod=*/(short)0, c, /*reuse_a=*/false, /*reuse_b=*/false);
}

// A fragment (16x32, row-major A, lane holds row m): per ISA layout,
// lane<16  -> K = {k0+0..7, k0+16..23}
// lane>=16 -> K = {k0+8..15, k0+24..31}
static __device__ __forceinline__ v16bf load_a_frag(const bf16_t* __restrict__ row,
                                                    int k0, int halfSel) {
  const bf16_t* p = row + k0 + halfSel * 8;
  v16bf a;
#pragma unroll
  for (int i = 0; i < 8; ++i) { a[i] = p[i]; a[i + 8] = p[i + 16]; }
  return a;
}

// B fragment from transposed weights Bt[N][K] (lane holds column n):
// lane<16 -> K = {k0+0..7, k0+16..23} ; lane>=16 -> K = {k0+8..15, k0+24..31}
static __device__ __forceinline__ v16bf load_b_frag(const bf16_t* __restrict__ row,
                                                    int k0, int halfSel) {
  const bf16_t* p = row + k0 + halfSel * 8;
  v16bf b;
#pragma unroll
  for (int i = 0; i < 8; ++i) { b[i] = p[i]; b[i + 8] = p[i + 16]; }
  return b;
}

static __device__ __forceinline__ v16bf zero_frag() {
  v16bf z;
#pragma unroll
  for (int i = 0; i < 16; ++i) z[i] = (bf16_t)0.0f;
  return z;
}

static __device__ __forceinline__ v8f zero_acc() { v8f z = {}; return z; }

// ------------------------------ generic GEMM -------------------------------
// One wave per 16x16 C tile (small seq-side GEMMs + attention).  The K loop
// runs two independent accumulator chains so adjacent WMMAs never have a
// D->C dependency on each other.

enum { EP_BIAS = 1, EP_RELU = 2, EP_SIG = 4, EP_RES = 8, EP_F32 = 16, EP_BF16 = 32 };

template <unsigned F>
__global__ __launch_bounds__(32)
void gemm16(const bf16_t* __restrict__ A, int lda, long long batchA,
            const bf16_t* __restrict__ Bt, int ldb, long long batchB,
            const float* __restrict__ bias,
            const float* __restrict__ resid, int ldres,
            float* __restrict__ Cf, bf16_t* __restrict__ Cbf,
            int ldc, long long batchC, int K) {
  const int lane = threadIdx.x;
  const int halfSel = lane >> 4, lr = lane & 15;
  const long long tM = (long long)blockIdx.y * 16;
  const int tN = blockIdx.x * 16;
  const long long z = blockIdx.z;

  const bf16_t* arow = A + z * batchA + (tM + lr) * (long long)lda;
  const bf16_t* brow = Bt + z * batchB + (long long)(tN + lr) * ldb;

  v8f acc0 = {}, acc1 = {};
  int k0 = 0;
  for (; k0 + 64 <= K; k0 += 64) {
    acc0 = wmma_bf16(load_a_frag(arow, k0, halfSel),
                     load_b_frag(brow, k0, halfSel), acc0);
    acc1 = wmma_bf16(load_a_frag(arow, k0 + 32, halfSel),
                     load_b_frag(brow, k0 + 32, halfSel), acc1);
  }
  if (k0 < K)
    acc0 = wmma_bf16(load_a_frag(arow, k0, halfSel),
                     load_b_frag(brow, k0, halfSel), acc0);

  const int cn = tN + lr;
  const float bv = (F & EP_BIAS) ? bias[cn] : 0.0f;
#pragma unroll
  for (int j = 0; j < 8; ++j) {
    long long cm = tM + j + halfSel * 8;
    float v = acc0[j] + acc1[j] + bv;
    if (F & EP_RES)  v += resid[cm * (long long)ldres + cn];
    if (F & EP_RELU) v = v > 0.0f ? v : 0.0f;
    if (F & EP_SIG)  v = 1.0f / (1.0f + __expf(-v));
    long long idx = z * batchC + cm * (long long)ldc + cn;
    if (F & EP_F32)  Cf[idx] = v;
    if (F & EP_BF16) Cbf[idx] = (bf16_t)v;
  }
}

// --------------- register-blocked implicit-GEMM conv (NHWC, CZ=128) --------
// One wave computes 32 positions x 128 output channels: 2 M-subtiles x
// 8 N-tiles = 16 independent accumulators.  ksize==1 degenerates to a plain
// [P,128] @ [128,128] GEMM (used for the s2p projection).
// act: [384*384][128] bf16, wt: [taps][co=128][ci=128] bf16 (pre-transposed).

template <unsigned F>
__global__ __launch_bounds__(32)
void conv_mn(const bf16_t* __restrict__ act, const bf16_t* __restrict__ wt,
             int ksize, int dil,
             const float* __restrict__ bias, const float* __restrict__ resid,
             float* __restrict__ Cf, bf16_t* __restrict__ Cbf) {
  const int lane = threadIdx.x;
  const int halfSel = lane >> 4, lr = lane & 15;
  const long long tM = (long long)blockIdx.y * 32;
  const int rad = ksize >> 1, taps = ksize * ksize;

  const long long m0 = tM + lr, m1 = tM + 16 + lr;
  const int yy0 = (int)(m0 / LSEQ), xx0 = (int)(m0 % LSEQ);
  const int yy1 = (int)(m1 / LSEQ), xx1 = (int)(m1 % LSEQ);

  v8f acc[2][8];
#pragma unroll
  for (int mt = 0; mt < 2; ++mt)
#pragma unroll
    for (int nt = 0; nt < 8; ++nt) acc[mt][nt] = zero_acc();

  for (int t = 0; t < taps; ++t) {
    const int dy = (t / ksize - rad) * dil;
    const int dx = (t % ksize - rad) * dil;
    const int ya = yy0 + dy, xa = xx0 + dx;
    const int yb = yy1 + dy, xb = xx1 + dx;
    const bool ok0 = ((unsigned)ya < (unsigned)LSEQ) && ((unsigned)xa < (unsigned)LSEQ);
    const bool ok1 = ((unsigned)yb < (unsigned)LSEQ) && ((unsigned)xb < (unsigned)LSEQ);
    const bf16_t* wrow = wt + (long long)t * CZd * CZd + (long long)lr * CZd;
    if (t + 1 < taps)  // prefetch next tap's weight block (global_prefetch_b8)
      __builtin_prefetch(wt + (long long)(t + 1) * CZd * CZd + (long long)lane * CZd * 4, 0, 3);
#pragma unroll
    for (int k0 = 0; k0 < CZd; k0 += 32) {
      v16bf a0, a1;
      if (ok0) a0 = load_a_frag(act + ((long long)ya * LSEQ + xa) * CZd, k0, halfSel);
      else     a0 = zero_frag();
      if (ok1) a1 = load_a_frag(act + ((long long)yb * LSEQ + xb) * CZd, k0, halfSel);
      else     a1 = zero_frag();
#pragma unroll
      for (int nt = 0; nt < 8; ++nt) {
        v16bf b = load_b_frag(wrow + (long long)nt * 16 * CZd, k0, halfSel);
        acc[0][nt] = wmma_bf16(a0, b, acc[0][nt]);
        acc[1][nt] = wmma_bf16(a1, b, acc[1][nt]);
      }
    }
  }

#pragma unroll
  for (int mt = 0; mt < 2; ++mt) {
#pragma unroll
    for (int nt = 0; nt < 8; ++nt) {
      const int n = nt * 16 + lr;
      const float bv = (F & EP_BIAS) ? bias[n] : 0.0f;
#pragma unroll
      for (int j = 0; j < 8; ++j) {
        long long cm = tM + mt * 16 + j + halfSel * 8;
        float v = acc[mt][nt][j] + bv;
        if (F & EP_RES)  v += resid[cm * CZd + n];
        if (F & EP_RELU) v = v > 0.0f ? v : 0.0f;
        if (F & EP_F32)  Cf[cm * CZd + n] = v;
        if (F & EP_BF16) Cbf[cm * CZd + n] = (bf16_t)v;
      }
    }
  }
}

// ------------------- fused pair MLP (LN input precomputed) -----------------
// one wave per 16 rows: hid = relu(A@W1t + b1) staged bf16 in LDS (16x512),
// out = hid@W2t + b2 + resid.  N-tiles processed in pairs -> independent
// WMMA chains + A-fragment reuse.

__global__ __launch_bounds__(32)
void pair_mlp(const bf16_t* __restrict__ hbf, const bf16_t* __restrict__ w1t,
              const float* __restrict__ b1, const bf16_t* __restrict__ w2t,
              const float* __restrict__ b2, const float* __restrict__ resid,
              float* __restrict__ out) {
  __shared__ bf16_t hid[16 * 512];
  const int lane = threadIdx.x;
  const int halfSel = lane >> 4, lr = lane & 15;
  const long long tM = (long long)blockIdx.x * 16;

  // preload A fragments (K = 128 -> 4 frags)
  v16bf afr[4];
  {
    const bf16_t* arow = hbf + (tM + lr) * (long long)CZd;
#pragma unroll
    for (int kk = 0; kk < 4; ++kk) afr[kk] = load_a_frag(arow, kk * 32, halfSel);
  }

  // stage 1: 16x512 hidden, two N-tiles in flight
  for (int nt = 0; nt < 32; nt += 2) {
    const int n0 = nt * 16 + lr, n1 = n0 + 16;
    const bf16_t* brow0 = w1t + (long long)n0 * CZd;
    const bf16_t* brow1 = w1t + (long long)n1 * CZd;
    v8f acc0 = {}, acc1 = {};
#pragma unroll
    for (int kk = 0; kk < 4; ++kk) {
      acc0 = wmma_bf16(afr[kk], load_b_frag(brow0, kk * 32, halfSel), acc0);
      acc1 = wmma_bf16(afr[kk], load_b_frag(brow1, kk * 32, halfSel), acc1);
    }
    const float bv0 = b1[n0], bv1 = b1[n1];
#pragma unroll
    for (int j = 0; j < 8; ++j) {
      const int mm = j + halfSel * 8;
      float v0 = acc0[j] + bv0;
      float v1 = acc1[j] + bv1;
      hid[mm * 512 + n0] = (bf16_t)(v0 > 0.0f ? v0 : 0.0f);
      hid[mm * 512 + n1] = (bf16_t)(v1 > 0.0f ? v1 : 0.0f);
    }
  }
  __syncthreads();

  // stage 2: 16x128 output, K = 512, two N-tiles in flight
  for (int nt = 0; nt < 8; nt += 2) {
    const int n0 = nt * 16 + lr, n1 = n0 + 16;
    const bf16_t* brow0 = w2t + (long long)n0 * 512;
    const bf16_t* brow1 = w2t + (long long)n1 * 512;
    v8f acc0 = {}, acc1 = {};
    for (int k0 = 0; k0 < 512; k0 += 32) {
      v16bf a = load_a_frag(&hid[lr * 512], k0, halfSel);
      acc0 = wmma_bf16(a, load_b_frag(brow0, k0, halfSel), acc0);
      acc1 = wmma_bf16(a, load_b_frag(brow1, k0, halfSel), acc1);
    }
    const float bv0 = b2[n0], bv1 = b2[n1];
#pragma unroll
    for (int j = 0; j < 8; ++j) {
      long long mm = tM + j + halfSel * 8;
      out[mm * CZd + n0] = acc0[j] + bv0 + resid[mm * CZd + n0];
      out[mm * CZd + n1] = acc1[j] + bv1 + resid[mm * CZd + n1];
    }
  }
}

// ------------------------------- LN / softmax ------------------------------

__global__ __launch_bounds__(32)
void ln_rows_384(const float* __restrict__ x, const float* __restrict__ g,
                 const float* __restrict__ b, bf16_t* __restrict__ out) {
  const int row = blockIdx.x, lane = threadIdx.x;
  const float* xr = x + (long long)row * CSd;
  float v[12], s = 0.f, ss = 0.f;
#pragma unroll
  for (int j = 0; j < 12; ++j) { float t = xr[lane + j * 32]; v[j] = t; s += t; ss += t * t; }
#pragma unroll
  for (int o = 16; o > 0; o >>= 1) { s += __shfl_xor(s, o, 32); ss += __shfl_xor(ss, o, 32); }
  const float mean = s * (1.0f / CSd);
  const float inv = rsqrtf(ss * (1.0f / CSd) - mean * mean + 1e-5f);
  bf16_t* orow = out + (long long)row * CSd;
#pragma unroll
  for (int j = 0; j < 12; ++j) {
    const int c = lane + j * 32;
    orow[c] = (bf16_t)(((v[j] - mean) * inv) * g[c] + b[c]);
  }
}

__global__ __launch_bounds__(32)
void ln_rows_128(const float* __restrict__ x, const float* __restrict__ g,
                 const float* __restrict__ b, bf16_t* __restrict__ out) {
  const long long row = blockIdx.x;
  const int lane = threadIdx.x;
  const float* xr = x + row * CZd;
  float v[4], s = 0.f, ss = 0.f;
#pragma unroll
  for (int j = 0; j < 4; ++j) { float t = xr[lane * 4 + j]; v[j] = t; s += t; ss += t * t; }
#pragma unroll
  for (int o = 16; o > 0; o >>= 1) { s += __shfl_xor(s, o, 32); ss += __shfl_xor(ss, o, 32); }
  const float mean = s * (1.0f / CZd);
  const float inv = rsqrtf(ss * (1.0f / CZd) - mean * mean + 1e-5f);
  bf16_t* orow = out + row * CZd;
#pragma unroll
  for (int j = 0; j < 4; ++j) {
    const int c = lane * 4 + j;
    orow[c] = (bf16_t)(((v[j] - mean) * inv) * g[c] + b[c]);
  }
}

// LN(pair) fused with pair->seq bias projection (CZ=128 -> NH=8)
__global__ __launch_bounds__(32)
void ln_pair_p2s(const float* __restrict__ pair, const float* __restrict__ g,
                 const float* __restrict__ b, const float* __restrict__ w,
                 float* __restrict__ biasOut) {
  const long long p = blockIdx.x;
  const int lane = threadIdx.x;
  const float* xr = pair + p * CZd;
  float v[4], s = 0.f, ss = 0.f;
#pragma unroll
  for (int j = 0; j < 4; ++j) { float t = xr[lane * 4 + j]; v[j] = t; s += t; ss += t * t; }
#pragma unroll
  for (int o = 16; o > 0; o >>= 1) { s += __shfl_xor(s, o, 32); ss += __shfl_xor(ss, o, 32); }
  const float mean = s * (1.0f / CZd);
  const float inv = rsqrtf(ss * (1.0f / CZd) - mean * mean + 1e-5f);
  float acc[NH];
#pragma unroll
  for (int h = 0; h < NH; ++h) acc[h] = 0.f;
#pragma unroll
  for (int j = 0; j < 4; ++j) {
    const int c = lane * 4 + j;
    const float nv = ((v[j] - mean) * inv) * g[c] + b[c];
#pragma unroll
    for (int h = 0; h < NH; ++h) acc[h] += nv * w[c * NH + h];
  }
#pragma unroll
  for (int o = 16; o > 0; o >>= 1)
#pragma unroll
    for (int h = 0; h < NH; ++h) acc[h] += __shfl_xor(acc[h], o, 32);
  if (lane == 0) {
#pragma unroll
    for (int h = 0; h < NH; ++h) biasOut[p * NH + h] = acc[h];
  }
}

// softmax over attention rows, pair bias fused in
__global__ __launch_bounds__(32)
void softmax_rows(const float* __restrict__ scores, const float* __restrict__ bias,
                  bf16_t* __restrict__ out) {
  const int h = blockIdx.y, i = blockIdx.x, lane = threadIdx.x;
  const float* srow = scores + ((long long)h * LSEQ + i) * LSEQ;
  float v[12], mx = -3.0e38f;
#pragma unroll
  for (int j = 0; j < 12; ++j) {
    const int col = lane + j * 32;
    float t = srow[col] + bias[((long long)i * LSEQ + col) * NH + h];
    v[j] = t; mx = fmaxf(mx, t);
  }
#pragma unroll
  for (int o = 16; o > 0; o >>= 1) mx = fmaxf(mx, __shfl_xor(mx, o, 32));
  float s = 0.f;
#pragma unroll
  for (int j = 0; j < 12; ++j) { v[j] = __expf(v[j] - mx); s += v[j]; }
#pragma unroll
  for (int o = 16; o > 0; o >>= 1) s += __shfl_xor(s, o, 32);
  const float r = 1.0f / s;
  bf16_t* orow = out + ((long long)h * LSEQ + i) * LSEQ;
#pragma unroll
  for (int j = 0; j < 12; ++j) orow[lane + j * 32] = (bf16_t)(v[j] * r);
}

// ------------------------------ BN (per channel) ---------------------------

__global__ __launch_bounds__(256)
void bn_stats(const float* __restrict__ x, const float* __restrict__ g,
              const float* __restrict__ b, float* __restrict__ ss_out) {
  const int c = blockIdx.x;
  __shared__ float s1[256], s2[256];
  float s = 0.f, sq = 0.f;
  for (long long p = threadIdx.x; p < (long long)PAIRN; p += 256) {
    const float t = x[p * CZd + c];
    s += t; sq += t * t;
  }
  s1[threadIdx.x] = s; s2[threadIdx.x] = sq;
  __syncthreads();
  for (int o = 128; o > 0; o >>= 1) {
    if ((int)threadIdx.x < o) { s1[threadIdx.x] += s1[threadIdx.x + o]; s2[threadIdx.x] += s2[threadIdx.x + o]; }
    __syncthreads();
  }
  if (threadIdx.x == 0) {
    const float mean = s1[0] / (float)PAIRN;
    const float var = s2[0] / (float)PAIRN - mean * mean;
    const float sc = g[c] * rsqrtf(var + 1e-5f);
    ss_out[c] = sc;
    ss_out[CZd + c] = b[c] - mean * sc;
  }
}

// ------------------------------ elementwise --------------------------------

__global__ void bn_relu_cvt(const float* __restrict__ x, const float* __restrict__ ss,
                            bf16_t* __restrict__ out, long long n) {
  const long long i = (long long)blockIdx.x * blockDim.x + threadIdx.x;
  if (i >= n) return;
  const int c = (int)(i & (CZd - 1));
  const float t = x[i] * ss[c] + ss[CZd + c];
  out[i] = (bf16_t)(t > 0.f ? t : 0.f);
}

__global__ void cvt_bf16(const float* __restrict__ x, bf16_t* __restrict__ out, long long n) {
  const long long i = (long long)blockIdx.x * blockDim.x + threadIdx.x;
  if (i < n) out[i] = (bf16_t)x[i];
}

__global__ void gate_mul(const float* __restrict__ g, const float* __restrict__ o,
                         bf16_t* __restrict__ out, long long n) {
  const long long i = (long long)blockIdx.x * blockDim.x + threadIdx.x;
  if (i < n) out[i] = (bf16_t)(g[i] * o[i]);
}

// prod/diff outer features: f[i,j,c] = (c<64) ? qh[j,c]*kh[i,c] : qh[j,c-64]-kh[i,c-64]
__global__ void make_pd(const float* __restrict__ s, bf16_t* __restrict__ pd) {
  const long long id = (long long)blockIdx.x * blockDim.x + threadIdx.x;
  if (id >= (long long)PAIRN * CZd) return;
  const int c = (int)(id & (CZd - 1));
  const long long p = id >> 7;
  const int j = (int)(p % LSEQ), i = (int)(p / LSEQ);
  const int cc = (c < 64) ? c : (c - 64);
  const float qv = s[(long long)j * CZd + cc];
  const float kv = s[(long long)i * CZd + 64 + cc];
  pd[id] = (bf16_t)((c < 64) ? (qv * kv) : (qv - kv));
}

// split qkv: q scaled by HW^-0.5, q/k padded to 64, v stored transposed per head
__global__ void split_qkv(const float* __restrict__ qkv, bf16_t* __restrict__ qb,
                          bf16_t* __restrict__ kb, bf16_t* __restrict__ vtb) {
  const long long id = (long long)blockIdx.x * blockDim.x + threadIdx.x;
  if (id >= (long long)NH * LSEQ * HWP) return;
  const int c = (int)(id & (HWP - 1));
  const long long t = id >> 6;
  const int m = (int)(t % LSEQ);
  const int h = (int)(t / LSEQ);
  const long long base = (long long)m * (3 * CSd) + (long long)h * (3 * HWd);
  const float qv = (c < HWd) ? qkv[base + c] * 0.14433756729740643f : 0.0f;
  const float kv = (c < HWd) ? qkv[base + HWd + c] : 0.0f;
  const long long o = ((long long)h * LSEQ + m) * HWP + c;
  qb[o] = (bf16_t)qv;
  kb[o] = (bf16_t)kv;
  if (c < HWd) vtb[((long long)h * HWd + c) * LSEQ + m] = (bf16_t)qkv[base + 2 * HWd + c];
}

// transpose + convert weights: dst[n*Kd+k] = (bf16)src[k*Nd+n], batched over grid.y
__global__ void transpose_cvt(const float* __restrict__ src, bf16_t* __restrict__ dst,
                              int Kd, int Nd) {
  const long long total = (long long)Kd * Nd;
  const long long base = (long long)blockIdx.y * total;
  for (long long i = (long long)blockIdx.x * blockDim.x + threadIdx.x; i < total;
       i += (long long)gridDim.x * blockDim.x) {
    const long long n = i / Kd, k = i - n * Kd;
    dst[base + i] = (bf16_t)src[base + k * Nd + n];
  }
}

// ------------------------------- host driver -------------------------------

extern "C" void kernel_launch(void* const* d_in, const int* in_sizes, int n_in,
                              void* d_out, int out_size, void* d_ws, size_t ws_size,
                              hipStream_t stream) {
  (void)in_sizes; (void)n_in; (void)out_size; (void)ws_size;

  const float* seq       = (const float*)d_in[0];
  const float* pair      = (const float*)d_in[1];
  const float* ln0_g     = (const float*)d_in[2];
  const float* ln0_b     = (const float*)d_in[3];
  const float* p2s_ln_g  = (const float*)d_in[4];
  const float* p2s_ln_b  = (const float*)d_in[5];
  const float* p2s_w     = (const float*)d_in[6];
  const float* attn_qkv_w= (const float*)d_in[7];
  const float* attn_o_w  = (const float*)d_in[8];
  const float* attn_o_b  = (const float*)d_in[9];
  const float* attn_g_w  = (const float*)d_in[10];
  const float* attn_g_b  = (const float*)d_in[11];
  const float* mlps_ln_g = (const float*)d_in[12];
  const float* mlps_ln_b = (const float*)d_in[13];
  const float* mlps_w1   = (const float*)d_in[14];
  const float* mlps_b1   = (const float*)d_in[15];
  const float* mlps_w2   = (const float*)d_in[16];
  const float* mlps_b2   = (const float*)d_in[17];
  const float* s2p_ln_g  = (const float*)d_in[18];
  const float* s2p_ln_b  = (const float*)d_in[19];
  const float* s2p_proj_w= (const float*)d_in[20];
  const float* s2p_proj_b= (const float*)d_in[21];
  const float* s2p_o_w   = (const float*)d_in[22];
  const float* s2p_o_b   = (const float*)d_in[23];
  const float* rb0_bn_g  = (const float*)d_in[24];
  const float* rb0_bn_b  = (const float*)d_in[25];
  const float* rb0_c1_w  = (const float*)d_in[26];
  const float* rb0_c2_w  = (const float*)d_in[27];
  const float* rb1_bn_g  = (const float*)d_in[28];
  const float* rb1_bn_b  = (const float*)d_in[29];
  const float* rb1_c1_w  = (const float*)d_in[30];
  const float* rb1_c2_w  = (const float*)d_in[31];
  const float* conv_out_w= (const float*)d_in[32];
  const float* conv_out_b= (const float*)d_in[33];
  const float* mlpp_ln_g = (const float*)d_in[34];
  const float* mlpp_ln_b = (const float*)d_in[35];
  const float* mlpp_w1   = (const float*)d_in[36];
  const float* mlpp_b1   = (const float*)d_in[37];
  const float* mlpp_w2   = (const float*)d_in[38];
  const float* mlpp_b2   = (const float*)d_in[39];

  float* out_seq  = (float*)d_out;            // 147456
  float* out_pair = (float*)d_out + PAIRN;    // 18874368

  // ---- workspace carve-out ----
  char* wsp = (char*)d_ws;
  size_t off = 0;
  auto carve = [&](size_t bytes) -> char* {
    char* p = wsp + off;
    off += (bytes + 255) & ~(size_t)255;
    return p;
  };

  bf16_t* wqkv_t  = (bf16_t*)carve((size_t)1152 * 384 * 2);
  bf16_t* wg_t    = (bf16_t*)carve((size_t)384 * 384 * 2);
  bf16_t* wo_t    = (bf16_t*)carve((size_t)384 * 384 * 2);
  bf16_t* w1_t    = (bf16_t*)carve((size_t)1536 * 384 * 2);
  bf16_t* w2_t    = (bf16_t*)carve((size_t)384 * 1536 * 2);
  bf16_t* sproj_t = (bf16_t*)carve((size_t)128 * 384 * 2);
  bf16_t* so_t    = (bf16_t*)carve((size_t)128 * 128 * 2);
  bf16_t* rb0c1_t = (bf16_t*)carve((size_t)9 * 128 * 128 * 2);
  bf16_t* rb0c2_t = (bf16_t*)carve((size_t)9 * 128 * 128 * 2);
  bf16_t* rb1c1_t = (bf16_t*)carve((size_t)9 * 128 * 128 * 2);
  bf16_t* rb1c2_t = (bf16_t*)carve((size_t)9 * 128 * 128 * 2);
  bf16_t* cout_t  = (bf16_t*)carve((size_t)49 * 128 * 128 * 2);
  bf16_t* mw1_t   = (bf16_t*)carve((size_t)512 * 128 * 2);
  bf16_t* mw2_t   = (bf16_t*)carve((size_t)128 * 512 * 2);

  float*  biasA   = (float*)carve((size_t)PAIRN * NH * 4);
  bf16_t* y_bf    = (bf16_t*)carve((size_t)384 * 384 * 2);
  float*  qkv_f   = (float*)carve((size_t)384 * 1152 * 4);
  bf16_t* q_bf    = (bf16_t*)carve((size_t)NH * 384 * HWP * 2);
  bf16_t* k_bf    = (bf16_t*)carve((size_t)NH * 384 * HWP * 2);
  bf16_t* vT_bf   = (bf16_t*)carve((size_t)NH * HWd * 384 * 2);
  float*  scores  = (float*)carve((size_t)NH * 384 * 384 * 4);
  bf16_t* a_bf    = (bf16_t*)carve((size_t)NH * 384 * 384 * 2);
  float*  o_f     = (float*)carve((size_t)384 * 384 * 4);
  float*  gate_f  = (float*)carve((size_t)384 * 384 * 4);
  bf16_t* og_bf   = (bf16_t*)carve((size_t)384 * 384 * 2);
  float*  seq1_f  = (float*)carve((size_t)384 * 384 * 4);
  bf16_t* h_bf    = (bf16_t*)carve((size_t)384 * 384 * 2);
  bf16_t* mlph_bf = (bf16_t*)carve((size_t)384 * 1536 * 2);
  bf16_t* sln_bf  = (bf16_t*)carve((size_t)384 * 384 * 2);
  float*  s_f     = (float*)carve((size_t)384 * 128 * 4);
  bf16_t* pd_bf   = (bf16_t*)carve((size_t)PAIRN * CZd * 2);
  float*  pairx_f = (float*)carve((size_t)PAIRN * CZd * 4);
  bf16_t* act_bf  = (bf16_t*)carve((size_t)PAIRN * CZd * 2);
  bf16_t* act2_bf = (bf16_t*)carve((size_t)PAIRN * CZd * 2);
  float*  pairy_f = (float*)carve((size_t)PAIRN * CZd * 4);
  float*  bnss    = (float*)carve((size_t)2 * CZd * 4);

  const long long NPC = (long long)PAIRN * CZd;   // 18.9M
  auto blocks1d = [](long long n) { return dim3((unsigned)((n + 255) / 256)); };
  const dim3 convGrid(1, PAIRN / 32);             // 4608 waves of 32x128 outputs

  // ---- 0. weight transpose + bf16 conversion ----
  transpose_cvt<<<blocks1d(384LL * 1152), 256, 0, stream>>>(attn_qkv_w, wqkv_t, 384, 1152);
  transpose_cvt<<<blocks1d(384LL * 384),  256, 0, stream>>>(attn_g_w,  wg_t,  384, 384);
  transpose_cvt<<<blocks1d(384LL * 384),  256, 0, stream>>>(attn_o_w,  wo_t,  384, 384);
  transpose_cvt<<<blocks1d(384LL * 1536), 256, 0, stream>>>(mlps_w1,   w1_t,  384, 1536);
  transpose_cvt<<<blocks1d(1536LL * 384), 256, 0, stream>>>(mlps_w2,   w2_t,  1536, 384);
  transpose_cvt<<<blocks1d(384LL * 128),  256, 0, stream>>>(s2p_proj_w, sproj_t, 384, 128);
  transpose_cvt<<<blocks1d(128LL * 128),  256, 0, stream>>>(s2p_o_w,   so_t,  128, 128);
  transpose_cvt<<<dim3(64, 9),  256, 0, stream>>>(rb0_c1_w, rb0c1_t, 128, 128);
  transpose_cvt<<<dim3(64, 9),  256, 0, stream>>>(rb0_c2_w, rb0c2_t, 128, 128);
  transpose_cvt<<<dim3(64, 9),  256, 0, stream>>>(rb1_c1_w, rb1c1_t, 128, 128);
  transpose_cvt<<<dim3(64, 9),  256, 0, stream>>>(rb1_c2_w, rb1c2_t, 128, 128);
  transpose_cvt<<<dim3(64, 49), 256, 0, stream>>>(conv_out_w, cout_t, 128, 128);
  transpose_cvt<<<blocks1d(128LL * 512), 256, 0, stream>>>(mlpp_w1, mw1_t, 128, 512);
  transpose_cvt<<<blocks1d(512LL * 128), 256, 0, stream>>>(mlpp_w2, mw2_t, 512, 128);

  // ---- 1. attention bias from LN(pair) (original pair) ----
  ln_pair_p2s<<<PAIRN, 32, 0, stream>>>(pair, p2s_ln_g, p2s_ln_b, p2s_w, biasA);

  // ---- 2. sequence attention path ----
  ln_rows_384<<<384, 32, 0, stream>>>(seq, ln0_g, ln0_b, y_bf);

  gemm16<EP_F32><<<dim3(72, 24, 1), 32, 0, stream>>>(
      y_bf, 384, 0, wqkv_t, 384, 0, nullptr, nullptr, 0,
      qkv_f, nullptr, 1152, 0, 384);

  split_qkv<<<blocks1d((long long)NH * 384 * HWP), 256, 0, stream>>>(qkv_f, q_bf, k_bf, vT_bf);

  // scores[h] = q[h] @ k[h]^T   (batched over heads)
  gemm16<EP_F32><<<dim3(24, 24, 8), 32, 0, stream>>>(
      q_bf, HWP, (long long)384 * HWP, k_bf, HWP, (long long)384 * HWP,
      nullptr, nullptr, 0, scores, nullptr, 384, (long long)384 * 384, HWP);

  softmax_rows<<<dim3(384, 8), 32, 0, stream>>>(scores, biasA, a_bf);

  // o[:, h*48+c] = a[h] @ v[h]   (batchC is a column offset of 48)
  gemm16<EP_F32><<<dim3(3, 24, 8), 32, 0, stream>>>(
      a_bf, 384, (long long)384 * 384, vT_bf, 384, (long long)HWd * 384,
      nullptr, nullptr, 0, o_f, nullptr, 384, (long long)HWd, 384);

  gemm16<EP_BIAS | EP_SIG | EP_F32><<<dim3(24, 24, 1), 32, 0, stream>>>(
      y_bf, 384, 0, wg_t, 384, 0, attn_g_b, nullptr, 0,
      gate_f, nullptr, 384, 0, 384);

  gate_mul<<<blocks1d(384LL * 384), 256, 0, stream>>>(gate_f, o_f, og_bf, 384LL * 384);

  gemm16<EP_BIAS | EP_RES | EP_F32><<<dim3(24, 24, 1), 32, 0, stream>>>(
      og_bf, 384, 0, wo_t, 384, 0, attn_o_b, seq, 384,
      seq1_f, nullptr, 384, 0, 384);

  // ---- 3. sequence MLP ----
  ln_rows_384<<<384, 32, 0, stream>>>(seq1_f, mlps_ln_g, mlps_ln_b, h_bf);

  gemm16<EP_BIAS | EP_RELU | EP_BF16><<<dim3(96, 24, 1), 32, 0, stream>>>(
      h_bf, 384, 0, w1_t, 384, 0, mlps_b1, nullptr, 0,
      nullptr, mlph_bf, 1536, 0, 384);

  gemm16<EP_BIAS | EP_RES | EP_F32><<<dim3(24, 24, 1), 32, 0, stream>>>(
      mlph_bf, 1536, 0, w2_t, 1536, 0, mlps_b2, seq1_f, 384,
      out_seq, nullptr, 384, 0, 1536);                       // final seq -> d_out

  // ---- 4. seq -> pair outer product ----
  ln_rows_384<<<384, 32, 0, stream>>>(out_seq, s2p_ln_g, s2p_ln_b, sln_bf);

  gemm16<EP_BIAS | EP_F32><<<dim3(8, 24, 1), 32, 0, stream>>>(
      sln_bf, 384, 0, sproj_t, 384, 0, s2p_proj_b, nullptr, 0,
      s_f, nullptr, 128, 0, 384);

  make_pd<<<blocks1d(NPC), 256, 0, stream>>>(s_f, pd_bf);

  // [P,128] @ [128,128] + bias + pair residual, via 1-tap conv_mn
  conv_mn<EP_BIAS | EP_RES | EP_F32><<<convGrid, 32, 0, stream>>>(
      pd_bf, so_t, 1, 1, s2p_o_b, pair, pairx_f, nullptr);

  // ---- 5. residual conv block 0 (dil 48, then 1) ----
  bn_stats<<<128, 256, 0, stream>>>(pairx_f, rb0_bn_g, rb0_bn_b, bnss);
  bn_relu_cvt<<<blocks1d(NPC), 256, 0, stream>>>(pairx_f, bnss, act_bf, NPC);
  conv_mn<EP_RELU | EP_BF16><<<convGrid, 32, 0, stream>>>(
      act_bf, rb0c1_t, 3, 48, nullptr, nullptr, nullptr, act2_bf);
  conv_mn<EP_RES | EP_F32><<<convGrid, 32, 0, stream>>>(
      act2_bf, rb0c2_t, 3, 1, nullptr, pairx_f, pairx_f, nullptr); // in-place residual

  // ---- 6. residual conv block 1 (dil 36, then 2) ----
  bn_stats<<<128, 256, 0, stream>>>(pairx_f, rb1_bn_g, rb1_bn_b, bnss);
  bn_relu_cvt<<<blocks1d(NPC), 256, 0, stream>>>(pairx_f, bnss, act_bf, NPC);
  conv_mn<EP_RELU | EP_BF16><<<convGrid, 32, 0, stream>>>(
      act_bf, rb1c1_t, 3, 36, nullptr, nullptr, nullptr, act2_bf);
  conv_mn<EP_RES | EP_F32><<<convGrid, 32, 0, stream>>>(
      act2_bf, rb1c2_t, 3, 2, nullptr, pairx_f, pairx_f, nullptr);

  // ---- 7. 7x7 output conv ----
  cvt_bf16<<<blocks1d(NPC), 256, 0, stream>>>(pairx_f, act_bf, NPC);
  conv_mn<EP_BIAS | EP_F32><<<convGrid, 32, 0, stream>>>(
      act_bf, cout_t, 7, 1, conv_out_b, nullptr, pairy_f, nullptr);

  // ---- 8. pair MLP (fused, LDS-staged hidden) -> d_out ----
  ln_rows_128<<<PAIRN, 32, 0, stream>>>(pairy_f, mlpp_ln_g, mlpp_ln_b, act_bf);
  pair_mlp<<<9216, 32, 0, stream>>>(act_bf, mw1_t, mlpp_b1, mw2_t, mlpp_b2,
                                    pairy_f, out_pair);
}
